// CapsNet_29987461661058
// MI455X (gfx1250) — compile-verified, matrix operand order checked by reading, not run on hardware
//
#include <hip/hip_runtime.h>
#include <hip/hip_bf16.h>

typedef __attribute__((ext_vector_type(16))) _Float16 v16h;
typedef __attribute__((ext_vector_type(8)))  _Float16 v8h;
typedef __attribute__((ext_vector_type(8)))  float    v8f;
typedef __attribute__((ext_vector_type(4)))  unsigned int v4u;
typedef __attribute__((ext_vector_type(8)))  int      v8i;
typedef __attribute__((ext_vector_type(4)))  int      v4i;

union ABFrag { v16h v; v8h h[2]; };

#if defined(__gfx1250__) && __has_builtin(__builtin_amdgcn_tensor_load_to_lds)
#define HAVE_TDM 1
#else
#define HAVE_TDM 0
#endif

// ---------------------------------------------------------------------------
// zero-init kernel (b_logits start at 0; ws is poisoned by harness)
// ---------------------------------------------------------------------------
__global__ void k_zero(float* __restrict__ p, int n) {
    int i = blockIdx.x * blockDim.x + threadIdx.x;
    if (i < n) p[i] = 0.0f;
}

// ---------------------------------------------------------------------------
// conv2 weight relayout: w2[n][c][ky][kx] f32 -> wB[ky*9+kx][n][c] f16
// so B-fragment loads are channel-contiguous 16B vectors.
// ---------------------------------------------------------------------------
__global__ void k_w2half(const float* __restrict__ w2, _Float16* __restrict__ wB) {
    int idx  = blockIdx.x * 256 + threadIdx.x;   // 81*256*256 = 5,308,416 exact
    int kykx = idx >> 16;
    int n    = (idx >> 8) & 255;
    int c    = idx & 255;
    wB[idx] = (_Float16)w2[(n * 256 + c) * 81 + kykx];
}

// ---------------------------------------------------------------------------
// conv1: [256,1,28,28] * [256,1,9,9] -> relu -> fp16 NHWC act[b][y][x][co]
// One block per output pixel (b,y,x); 256 threads = 256 output channels.
// ---------------------------------------------------------------------------
__global__ void k_conv1(const float* __restrict__ x, const float* __restrict__ w1,
                        const float* __restrict__ b1, _Float16* __restrict__ act) {
    __shared__ float patch[81];
    int bx  = blockIdx.x;          // b*400 + y*20 + x
    int b   = bx / 400;
    int rem = bx % 400;
    int oy  = rem / 20, ox = rem % 20;
    int t   = threadIdx.x;
    if (t < 81) {
        int ky = t / 9, kx = t % 9;
        patch[t] = x[b * 784 + (oy + ky) * 28 + (ox + kx)];
    }
    __syncthreads();
    float acc = b1[t];
    const float* wp = w1 + t * 81;
    #pragma unroll 9
    for (int i = 0; i < 81; ++i) acc += patch[i] * wp[i];
    acc = fmaxf(acc, 0.0f);
    act[(size_t)bx * 256 + t] = (_Float16)acc;   // NHWC, channel-contiguous
}

// ---------------------------------------------------------------------------
// conv2 implicit GEMM via WMMA f16, 2x2 register blocking: each wave owns a
// 32x32 output tile (4 v8f accumulators), so 8 b128 loads feed 4 WMMAs
// (2 loads/WMMA instead of 4). M=9216 (b,oy,ox), N=256 (ch), K=256 per tap,
// 81 taps -> no div/mod in the K loop.
// A fragment (16-bit A 16x32 layout): lane = row, two contiguous 8-f16 runs
//   K = kh*8+{0..7} and 16+kh*8+{0..7}  (kh = lane>>4)  -> two b128 loads.
// ---------------------------------------------------------------------------
__global__ void __launch_bounds__(128)
k_conv2(const _Float16* __restrict__ act, const _Float16* __restrict__ wB,
        const float* __restrict__ b2, float* __restrict__ hc2) {
    int lane = threadIdx.x & 31;
    int wave = threadIdx.x >> 5;
    int tile = blockIdx.x * 4 + wave;      // 0..2303 wave tiles (288 M x 8 N)
    int nt   = tile & 7;                   // N block of 32
    int mt   = tile >> 3;                  // M block of 32
    int ln   = lane & 15;                  // A: row-in-subtile. B/D: col.
    int kh   = lane >> 4;

    const _Float16* pA[2];
    #pragma unroll
    for (int r = 0; r < 2; ++r) {
        int Mr = mt * 32 + r * 16 + ln;
        int bb = Mr / 36;
        int s  = Mr % 36;
        int oy = s / 6, ox = s % 6;
        pA[r] = act + (size_t)((bb * 20 + 2 * oy) * 20 + 2 * ox) * 256 + kh * 8;
    }
    const _Float16* pB[2];
    #pragma unroll
    for (int q = 0; q < 2; ++q)
        pB[q] = wB + (size_t)(nt * 32 + q * 16 + ln) * 256 + kh * 8;

    v8f acc00 = {0.f,0.f,0.f,0.f,0.f,0.f,0.f,0.f};
    v8f acc01 = acc00, acc10 = acc00, acc11 = acc00;

    for (int ky = 0; ky < 9; ++ky) {
        for (int kx = 0; kx < 9; ++kx) {
            size_t offA = (size_t)(ky * 20 + kx) * 256;
            size_t offB = (size_t)(ky * 9 + kx) * 65536;
            const _Float16* a0 = pA[0] + offA;
            const _Float16* a1 = pA[1] + offA;
            const _Float16* b0 = pB[0] + offB;
            const _Float16* b1 = pB[1] + offB;
            __builtin_prefetch((const void*)(a0 + 256), 0, 0);  // next tap
            #pragma unroll
            for (int cs = 0; cs < 8; ++cs) {                    // K = 256 / 32
                ABFrag fa0, fa1, fb0, fb1;
                fa0.h[0] = *(const v8h*)(a0 + cs * 32);
                fa0.h[1] = *(const v8h*)(a0 + cs * 32 + 16);
                fa1.h[0] = *(const v8h*)(a1 + cs * 32);
                fa1.h[1] = *(const v8h*)(a1 + cs * 32 + 16);
                fb0.h[0] = *(const v8h*)(b0 + cs * 32);
                fb0.h[1] = *(const v8h*)(b0 + cs * 32 + 16);
                fb1.h[0] = *(const v8h*)(b1 + cs * 32);
                fb1.h[1] = *(const v8h*)(b1 + cs * 32 + 16);
                acc00 = __builtin_amdgcn_wmma_f32_16x16x32_f16(
                    false, fa0.v, false, fb0.v, (short)0, acc00, false, false);
                acc01 = __builtin_amdgcn_wmma_f32_16x16x32_f16(
                    false, fa0.v, false, fb1.v, (short)0, acc01, false, false);
                acc10 = __builtin_amdgcn_wmma_f32_16x16x32_f16(
                    false, fa1.v, false, fb0.v, (short)0, acc10, false, false);
                acc11 = __builtin_amdgcn_wmma_f32_16x16x32_f16(
                    false, fa1.v, false, fb1.v, (short)0, acc11, false, false);
            }
        }
    }

    v8f accs[2][2] = {{acc00, acc01}, {acc10, acc11}};
    #pragma unroll
    for (int q = 0; q < 2; ++q) {
        int   ch   = nt * 32 + q * 16 + ln;
        float bias = b2[ch];
        #pragma unroll
        for (int r = 0; r < 2; ++r) {
            #pragma unroll
            for (int j = 0; j < 8; ++j) {  // D layout: VGPR j -> M = j + 8*kh
                int Mj = mt * 32 + r * 16 + j + 8 * kh;
                int bj = Mj / 36;
                int sj = Mj % 36;
                hc2[(size_t)bj * 9216 + ch * 36 + sj] = accs[r][q][j] + bias;
            }
        }
    }
}

// ---------------------------------------------------------------------------
// primary squash: u[b][ncap][d] = (|t|^2/(1+|t|^2)) * t   (no /|t|)
// write transposed xp[ncap][d][b] for the u_hat kernel (batch-contiguous).
// ---------------------------------------------------------------------------
__global__ void k_squash(const float* __restrict__ hc2, float* __restrict__ xp) {
    int ncap = blockIdx.x;      // 0..1151
    int b    = threadIdx.x;     // 0..255
    const float* p = hc2 + (size_t)b * 9216 + ncap * 8;
    float4 lo = *(const float4*)p;
    float4 hi = *(const float4*)(p + 4);
    float v[8] = {lo.x, lo.y, lo.z, lo.w, hi.x, hi.y, hi.z, hi.w};
    float sq = 0.f;
    #pragma unroll
    for (int d = 0; d < 8; ++d) sq += v[d] * v[d];
    float scale = sq / (1.f + sq);
    #pragma unroll
    for (int d = 0; d < 8; ++d)
        xp[(size_t)(ncap * 8 + d) * 256 + b] = scale * v[d];
}

// ---------------------------------------------------------------------------
// u_hat[c][n][i][b] = sum_d W[c][n][i][d] * xp[n][d][b]
// One block per (c,n). The 16x8 f32 W tile (512 B) is DMA'd into LDS with the
// Tensor Data Mover (1-D descriptor: data_size=4B, tile_dim0=128, type=2),
// issued by wave 0, completion via s_wait_tensorcnt then block barrier.
// Falls back to plain loads if the TDM builtin is unavailable.
// ---------------------------------------------------------------------------
__global__ void k_uhat(const float* __restrict__ W, const float* __restrict__ xp,
                       float* __restrict__ uhat) {
    __shared__ float Wl[128];
    int c = blockIdx.x / 1152;
    int n = blockIdx.x % 1152;
    int t = threadIdx.x;
#if HAVE_TDM
    if (t < 32) {
        unsigned long long ga =
            (unsigned long long)(const void*)(W + (size_t)(c * 1152 + n) * 128);
        // D# group 0: count=1, lds_addr=0 (Wl is the only LDS object),
        //             global_addr[56:0], type=2 ("image")
        v4u g0 = { 1u, 0u, (unsigned)(ga & 0xFFFFFFFFu),
                   (unsigned)((ga >> 32) & 0x1FFFFFFu) | (2u << 30) };
        // D# group 1: data_size=2 (4B); tensor_dim0=128; tensor_dim1=1;
        //             tile_dim0=128 (1-D tile); tensor_dim0_stride=128
        v8i g1 = { (int)(2u << 16),
                   (int)(128u << 16),      // tensor_dim0[15:0] @ bits 63:48
                   (int)(1u << 16),        // tensor_dim0 hi=0, tensor_dim1 lo=1
                   (int)(128u << 16),      // tensor_dim1 hi=0, tile_dim0=128
                   0,                      // tile_dim1=0 (unused), tile_dim2=0
                   128, 0, 0 };            // tensor_dim0_stride=128
        v4i gz = {0, 0, 0, 0};
#if __clang_major__ >= 23
        v8i gz8 = {0, 0, 0, 0, 0, 0, 0, 0};
        __builtin_amdgcn_tensor_load_to_lds(g0, g1, gz, gz, gz8, 0);
#else
        __builtin_amdgcn_tensor_load_to_lds(g0, g1, gz, gz, 0);
#endif
        __builtin_amdgcn_s_wait_tensorcnt((short)0);
        asm volatile("" ::: "memory");
    }
    __syncthreads();
#else
    if (t < 128) Wl[t] = W[(size_t)(c * 1152 + n) * 128 + t];
    __syncthreads();
#endif
    float xv[8];
    #pragma unroll
    for (int d = 0; d < 8; ++d) xv[d] = xp[(size_t)(n * 8 + d) * 256 + t];
    float* up = uhat + (size_t)(c * 1152 + n) * 16 * 256 + t;
    #pragma unroll
    for (int i = 0; i < 16; ++i) {
        float acc = 0.f;
        #pragma unroll
        for (int d = 0; d < 8; ++d) acc += Wl[i * 8 + d] * xv[d];
        up[(size_t)i * 256] = acc;
    }
}

// ---------------------------------------------------------------------------
// routing step A (per (c,i) block): softmax over n of blog[c][n][i],
// s[b] = (1/Z) * sum_n e[n]*u_hat[c][n][i][b], batch-axis squash, write outv.
// u_hat (188.7 MB) stays resident in the 192 MB L2 across iterations.
// ---------------------------------------------------------------------------
__global__ void k_route1(const float* __restrict__ blog, const float* __restrict__ uhat,
                         float* __restrict__ outv) {
    __shared__ float probs[1152];
    __shared__ float red[256];
    int c = blockIdx.x >> 4;
    int i = blockIdx.x & 15;
    int t = threadIdx.x;

    float lmax = -3.4e38f;
    for (int n = t; n < 1152; n += 256)
        lmax = fmaxf(lmax, blog[(size_t)(c * 1152 + n) * 16 + i]);
    red[t] = lmax; __syncthreads();
    for (int s2 = 128; s2 > 0; s2 >>= 1) {
        if (t < s2) red[t] = fmaxf(red[t], red[t + s2]);
        __syncthreads();
    }
    float mx = red[0]; __syncthreads();

    float lsum = 0.f;
    for (int n = t; n < 1152; n += 256) {
        float e = __expf(blog[(size_t)(c * 1152 + n) * 16 + i] - mx);
        probs[n] = e; lsum += e;
    }
    red[t] = lsum; __syncthreads();
    for (int s2 = 128; s2 > 0; s2 >>= 1) {
        if (t < s2) red[t] += red[t + s2];
        __syncthreads();
    }
    float inv = 1.f / red[0]; __syncthreads();

    const float* up = uhat + ((size_t)(c * 1152) * 16 + i) * 256 + t;  // b = t
    float acc = 0.f;
    for (int n = 0; n < 1152; ++n) acc += probs[n] * up[(size_t)n * 16 * 256];
    acc *= inv;

    red[t] = acc * acc; __syncthreads();
    for (int s2 = 128; s2 > 0; s2 >>= 1) {
        if (t < s2) red[t] += red[t + s2];
        __syncthreads();
    }
    float sq    = red[0];
    float scale = (sq / (1.f + sq)) * rsqrtf(sq);
    outv[(size_t)(c * 16 + i) * 256 + t] = scale * acc;
}

// ---------------------------------------------------------------------------
// routing step B: blog[c][n][i] += sum_b u_hat[c][n][i][b] * outv[c][i][b]
// One block per (c,n); 256 threads = 16 dims x 16 batch lanes.
// ---------------------------------------------------------------------------
__global__ void k_route2(const float* __restrict__ uhat, const float* __restrict__ outv,
                         float* __restrict__ blog) {
    __shared__ float red[256];
    int c  = blockIdx.x / 1152;
    int n  = blockIdx.x % 1152;
    int t  = threadIdx.x;
    int i  = t >> 4;
    int bl = t & 15;
    const float* up = uhat + ((size_t)(c * 1152 + n) * 16 + i) * 256;
    const float* op = outv + (size_t)(c * 16 + i) * 256;
    float p = 0.f;
    #pragma unroll
    for (int k = 0; k < 16; ++k) { int b = bl + k * 16; p += up[b] * op[b]; }
    red[t] = p; __syncthreads();
    if (t < 16) {
        float s = 0.f;
        #pragma unroll
        for (int j = 0; j < 16; ++j) s += red[t * 16 + j];
        blog[(size_t)(c * 1152 + n) * 16 + t] += s;
    }
}

// ---------------------------------------------------------------------------
// final: out[b][c] = sum_i outv[c][i][b]^2
// ---------------------------------------------------------------------------
__global__ void k_final(const float* __restrict__ outv, float* __restrict__ out) {
    int c = blockIdx.x;
    int b = threadIdx.x;
    float acc = 0.f;
    #pragma unroll
    for (int i = 0; i < 16; ++i) {
        float v = outv[(size_t)(c * 16 + i) * 256 + b];
        acc += v * v;
    }
    out[b * 10 + c] = acc;
}

// ---------------------------------------------------------------------------
extern "C" void kernel_launch(void* const* d_in, const int* in_sizes, int n_in,
                              void* d_out, int out_size, void* d_ws, size_t ws_size,
                              hipStream_t stream) {
    (void)in_sizes; (void)n_in; (void)out_size; (void)ws_size;
    const float* x  = (const float*)d_in[0];
    const float* w1 = (const float*)d_in[1];
    const float* b1 = (const float*)d_in[2];
    const float* w2 = (const float*)d_in[3];
    const float* b2 = (const float*)d_in[4];
    const float* W  = (const float*)d_in[5];

    char* ws = (char*)d_ws;
    // u_hat region [0, 188,743,680): act/wB/hc2 aliased inside (dead before
    // u_hat is written).
    float*    uhat = (float*)(ws + 0);
    _Float16* act  = (_Float16*)(ws + 0);          // 52,428,800 B
    _Float16* wB   = (_Float16*)(ws + 52428800);   // 10,616,832 B
    float*    hc2  = (float*)(ws + 63045632);      //  9,437,184 B
    float*    xp   = (float*)(ws + 188743680);     //  9,437,184 B
    float*    blog = (float*)(ws + 198180864);     //    737,280 B
    float*    outv = (float*)(ws + 198918144);     //    163,840 B
    float*    out  = (float*)d_out;

    k_zero  <<<720,    256, 0, stream>>>(blog, 184320);
    k_w2half<<<20736,  256, 0, stream>>>(w2, wB);
    k_conv1 <<<102400, 256, 0, stream>>>(x, w1, b1, act);
    k_conv2 <<<576,    128, 0, stream>>>(act, wB, b2, hc2);
    k_squash<<<1152,   256, 0, stream>>>(hc2, xp);
    k_uhat  <<<11520,  256, 0, stream>>>(W, xp, uhat);
    for (int it = 0; it < 3; ++it) {
        k_route1<<<160, 256, 0, stream>>>(blog, uhat, outv);
        if (it < 2) k_route2<<<11520, 256, 0, stream>>>(uhat, outv, blog);
    }
    k_final<<<10, 256, 0, stream>>>(outv, out);
}
